// CMAAA_29274497089816
// MI455X (gfx1250) — matrix operand order repaired and checked by reading, not compiled, not run on hardware
//
#include <hip/hip_runtime.h>
#include <hip/hip_bf16.h>
#include <math.h>

// ---------------- problem constants ----------------
#define DIMC   32
#define HEADS  8
#define HD     4          // head dim
#define KA     3
#define KK     9          // KA*KA
#define MS_C   8
#define BATCH  2
#define HH     256
#define WW     256
#define HW     65536      // HH*WW
#define CU     51         // unified input channels: x32 + cond8 + ms8 + lpan + pan + (pan-lpan)
#define PADW   258        // WW+2 (zero-padded)
#define KTOT   459        // CU*KK
#define KPAD   512        // K padded to power of two (zero weights/data beyond KTOT)
#define NPB    64         // pixels per conv block (same image row; W%64==0)
#define QKVC   128        // 32 q + 32 v_pan + 32 k_ms + 32 v_ms

typedef __attribute__((ext_vector_type(16))) __bf16 v16bf;
typedef __attribute__((ext_vector_type(8)))  __bf16 v8bf;
typedef __attribute__((ext_vector_type(8)))  float  v8f;

// chunk-level XOR swizzle for the conv LDS tile: 32B chunks, chunk ^= (pixel & 31).
// Keeps each 16-element fragment contiguous & 32B-aligned while spreading banks.
__device__ __forceinline__ int swz(int n, int K) {
    return n * KPAD + (((((K >> 4) ^ n) & 31)) << 4) + (K & 15);
}

// ---------------- kernel 0a: build padded unified input (bf16) ----------------
__global__ void __launch_bounds__(256)
prep_U_kernel(const float* __restrict__ x, const float* __restrict__ ms,
              const float* __restrict__ lpan, const float* __restrict__ pan,
              const float* __restrict__ sarr, __bf16* __restrict__ U, int total)
{
    int i = blockIdx.x * 256 + threadIdx.x;
    if (i >= total) return;
    const int perB = CU * PADW * PADW;
    int b  = i / perB;
    int r  = i - b * perB;
    int c  = r / (PADW * PADW);
    int pr = r - c * (PADW * PADW);
    int xp = pr / PADW;
    int yp = pr - xp * PADW;

    float val = 0.0f;
    if (xp >= 1 && xp <= HH && yp >= 1 && yp <= WW) {
        int xx = xp - 1, yy = yp - 1;
        int sp = (xx << 8) + yy;
        if (c < 32) {
            val = x[((size_t)(b * DIMC + c) << 16) + sp];
        } else if (c < 40) {           // cond = lpan*(1-s) + ms*s  (lpan tiled over MS_C)
            float sv = sarr[b];
            float lp = lpan[((size_t)b << 16) + sp];
            float m  = ms[((size_t)(b * MS_C + (c - 32)) << 16) + sp];
            val = lp * (1.0f - sv) + m * sv;
        } else if (c < 48) {           // ms
            val = ms[((size_t)(b * MS_C + (c - 40)) << 16) + sp];
        } else if (c == 48) {
            val = lpan[((size_t)b << 16) + sp];
        } else if (c == 49) {
            val = pan[((size_t)b << 16) + sp];
        } else {                        // pan - lpan
            val = pan[((size_t)b << 16) + sp] - lpan[((size_t)b << 16) + sp];
        }
    }
    U[i] = (__bf16)val;
}

// ---------------- kernel 0b: unified conv weights + proj weights (bf16) ----------------
__global__ void __launch_bounds__(256)
prep_W_kernel(const float* __restrict__ wq, const float* __restrict__ wvpan,
              const float* __restrict__ wkvms, const float* __restrict__ wpp,
              const float* __restrict__ wpm, __bf16* __restrict__ Wall,
              __bf16* __restrict__ PW)
{
    int i = blockIdx.x * 256 + threadIdx.x;
    if (i < QKVC * KPAD) {
        int o = i >> 9;          // KPAD == 512
        int K = i & (KPAD - 1);
        float val = 0.0f;
        if (K < KTOT) {
            int c = K / KK;
            int t = K - c * KK;
            if (o < 32) {                                  // q: [x(32), cond(8)] == U ch 0..39
                if (c < 40) val = wq[(o * 40 + c) * KK + t];
            } else if (o < 64) {                           // v_pan: [x(32), lpan, pan, pan-lpan]
                int oo = o - 32, ci = -1;
                if (c < 32)       ci = c;
                else if (c == 48) ci = 32;
                else if (c == 49) ci = 33;
                else if (c == 50) ci = 34;
                if (ci >= 0) val = wvpan[(oo * 35 + ci) * KK + t];
            } else {                                       // kv_ms: [x(32), ms(8)]
                int oo = o - 64, ci = -1;
                if (c < 32)                 ci = c;
                else if (c >= 40 && c < 48) ci = 32 + (c - 40);
                if (ci >= 0) val = wkvms[(oo * 40 + ci) * KK + t];
            }
        }
        Wall[i] = (__bf16)val;
    } else {
        int i2 = i - QKVC * KPAD;                          // 2*32*32 proj weights
        if (i2 < 2048) {
            int s = i2 >> 10;
            int r = i2 & 1023;
            PW[i2] = (__bf16)(s ? wpm[r] : wpp[r]);
        }
    }
}

// ---------------- kernel 1: fused conv3x3 implicit GEMM via WMMA bf16 ----------------
// One block = 64 consecutive pixels (same row). im2col tile staged once in LDS (XOR-
// swizzled), shared by all 8 waves. Each wave owns out-channel tile [16w,16w+16) and
// reuses every A fragment across 4 pixel sub-tiles (4 WMMAs per weight load).
__global__ void __launch_bounds__(256)
conv_gemm_kernel(const __bf16* __restrict__ U, const __bf16* __restrict__ Wall,
                 __bf16* __restrict__ qkv)
{
    __shared__ __bf16 sB[NPB * KPAD];       // 64 * 512 * 2B = 64 KB

    const int tid = threadIdx.x;
    const int P0  = blockIdx.x * NPB;       // 64 pixels, same row
    const int b   = P0 >> 16;
    const int rm  = P0 & (HW - 1);
    const int x   = rm >> 8;
    const int y0  = rm & 255;
    const __bf16* __restrict__ Ub = U + (size_t)b * CU * PADW * PADW;

    // cooperative im2col fill: e = K*64 + n -> coalesced 64-pixel runs per (c,tap)
    for (int e = tid; e < KPAD * NPB; e += 256) {
        int K = e >> 6;
        int n = e & (NPB - 1);
        __bf16 v = (__bf16)0.0f;
        if (K < KTOT) {
            int c  = K / KK;
            int t  = K - c * KK;
            int ty = t / KA;
            int tx = t - ty * KA;
            v = Ub[((size_t)c * PADW + (x + ty)) * PADW + (y0 + n + tx)];
        }
        sB[swz(n, K)] = v;
    }
    __syncthreads();

    const int wave = tid >> 5;
    const int lane = tid & 31;
    const int half = lane >> 4;
    const int l16  = lane & 15;
    const int m0   = wave * 16;
    const int row  = m0 + l16;

    const __bf16* __restrict__ Arow = Wall + (size_t)row * KPAD;

    v8f acc[4] = {};
    for (int kc = 0; kc < KPAD; kc += 32) {
        // A fragment (16-bit 16x32 layout): two contiguous 16B halves
        v8bf lo = *(const v8bf*)(Arow + kc + (half << 4));
        v8bf hi = *(const v8bf*)(Arow + kc + (half << 4) + 8);
        v16bf a = __builtin_shufflevector(lo, hi, 0, 1, 2, 3, 4, 5, 6, 7,
                                          8, 9, 10, 11, 12, 13, 14, 15);
        // reuse A across 4 pixel sub-tiles
#pragma unroll
        for (int st = 0; st < 4; ++st) {
            v16bf bm = *(const v16bf*)(sB + swz(st * 16 + l16, kc + (half << 4)));
            acc[st] = __builtin_amdgcn_wmma_f32_16x16x32_bf16(
                false, a, false, bm, (short)0, acc[st], false, false);
        }
    }
    // D/C layout: element r -> M = r + 8*half, N = l16
#pragma unroll
    for (int st = 0; st < 4; ++st) {
        int y = y0 + st * 16 + l16;
#pragma unroll
        for (int r = 0; r < 8; ++r) {
            int oc = m0 + r + (half << 3);
            qkv[(((size_t)b * QKVC + oc) << 16) + (x << 8) + y] = (__bf16)acc[st][r];
        }
    }
}

// ---------------- kernel 2: fused depthwise-9 + attention + softmax ----------------
// qkv sections: [0,32)=q  [32,64)=v_pan  [64,96)=k_ms  [96,128)=v_ms
__global__ void __launch_bounds__(256)
attn_kernel(const __bf16* __restrict__ qkv, const float* __restrict__ w_dep,
            const float* __restrict__ b_dep, __bf16* __restrict__ ao)
{
    __shared__ float sw[HD * KK * KK];   // 324
    __shared__ float sb[HD * KK];        // 36
    for (int j = threadIdx.x; j < HD * KK * KK + HD * KK; j += 256) {
        if (j < HD * KK * KK) sw[j] = w_dep[j];
        else                  sb[j - HD * KK * KK] = b_dep[j - HD * KK * KK];
    }
    __syncthreads();

    int i   = blockIdx.x * 256 + threadIdx.x;   // over B*HEADS*H*W
    int b   = i >> 19;
    int rem = i & ((1 << 19) - 1);
    int h   = rem >> 16;
    int p   = rem & (HW - 1);
    int x   = p >> 8;
    int y   = p & 255;

    const __bf16* __restrict__ base = qkv + ((size_t)b * QKVC << 16);
    auto ld = [&](int ch, int xx, int yy) -> float {
        return (xx >= 0 && xx < HH && yy >= 0 && yy < WW)
                   ? (float)base[((size_t)ch << 16) + (xx << 8) + yy] : 0.0f;
    };

    float q[HD];
#pragma unroll
    for (int d = 0; d < HD; ++d)
        q[d] = (float)base[((size_t)(h * HD + d) << 16) + (x << 8) + y] * 0.5f; // scale=hd^-0.5

#pragma unroll
    for (int s = 0; s < 2; ++s) {
        // ---- key neighborhoods (s=0: k_ms normal since k_pan:=k_ms; s=1: scrambled view) ----
        float nb[HD][KK];
#pragma unroll
        for (int d = 0; d < HD; ++d)
#pragma unroll
            for (int t = 0; t < KK; ++t) {
                int ty = t / KA, tx = t - ty * KA;
                int xx = x + ty - 1, yy = y + tx - 1;
                if (s == 0) {
                    nb[d][t] = ld(64 + h * HD + d, xx, yy);
                } else {
                    // k_ms_flat[b,h,d,xx,yy] = k_ms[b,h, xx&3, yy, (d<<6)+(xx>>2)]
                    float v = 0.0f;
                    if (xx >= 0 && xx < HH && yy >= 0 && yy < WW)
                        v = (float)base[((size_t)(64 + h * HD + (xx & 3)) << 16)
                                        + (yy << 8) + ((d << 6) + (xx >> 2))];
                    nb[d][t] = v;
                }
            }
        // ---- logits over 9 keys ----
        float lg[KK];
#pragma unroll
        for (int j = 0; j < KK; ++j) {
            float acc = 0.0f;
#pragma unroll
            for (int d = 0; d < HD; ++d) {
                float kv = sb[d * KK + j];
#pragma unroll
                for (int t = 0; t < KK; ++t)
                    kv += nb[d][t] * sw[(d * KK + j) * KK + t];
                acc += q[d] * kv;
            }
            lg[j] = acc;
        }
        // ---- softmax over k ----
        float mx = lg[0];
#pragma unroll
        for (int j = 1; j < KK; ++j) mx = fmaxf(mx, lg[j]);
        float sum = 0.0f;
#pragma unroll
        for (int j = 0; j < KK; ++j) { lg[j] = __expf(lg[j] - mx); sum += lg[j]; }
        float inv = 1.0f / sum;
#pragma unroll
        for (int j = 0; j < KK; ++j) lg[j] *= inv;
        // ---- value neighborhoods: s=0 -> v_pan, s=1 -> v_ms ----
        int vbase = s ? 96 : 32;
#pragma unroll
        for (int d = 0; d < HD; ++d)
#pragma unroll
            for (int t = 0; t < KK; ++t) {
                int ty = t / KA, tx = t - ty * KA;
                nb[d][t] = ld(vbase + h * HD + d, x + ty - 1, y + tx - 1);
            }
        // ---- weighted values ----
#pragma unroll
        for (int d = 0; d < HD; ++d) {
            float o = 0.0f;
#pragma unroll
            for (int j = 0; j < KK; ++j) {
                float vv = sb[d * KK + j];
#pragma unroll
                for (int t = 0; t < KK; ++t)
                    vv += nb[d][t] * sw[(d * KK + j) * KK + t];
                o += lg[j] * vv;
            }
            ao[((size_t)((b * 2 + s) * DIMC + h * HD + d) << 16) + (x << 8) + y] = (__bf16)o;
        }
    }
}

// ---------------- kernel 3: 1x1 projection GEMM via WMMA + bias ----------------
// One block = 32 pixels; ao tile transposed through LDS for contiguous fragments.
// Waves: nt2(pixel half) x s(branch) x mt(out-ch tile).
__global__ void __launch_bounds__(256)
proj_kernel(const __bf16* __restrict__ ao, const __bf16* __restrict__ pw,
            const float* __restrict__ bias_pan, const float* __restrict__ bias_ms,
            float* __restrict__ out)
{
    __shared__ __bf16 sT[2 * 2 * 16 * 40];   // [nt2][s][pixel n][K(32), stride 40]

    const int tid = threadIdx.x;
    const int P0  = blockIdx.x * 32;

    // cooperative transpose fill: coalesced 16-pixel runs per (nt2, s, channel K)
    for (int e = tid; e < 2048; e += 256) {
        int n   = e & 15;
        int K   = (e >> 4) & 31;
        int s   = (e >> 9) & 1;
        int nt2 = (e >> 10) & 1;
        int P   = P0 + nt2 * 16 + n;
        int b   = P >> 16;
        int rm  = P & (HW - 1);
        int x   = rm >> 8;
        int y   = rm & 255;
        sT[((nt2 * 2 + s) * 16 + n) * 40 + K] =
            ao[((size_t)((b * 2 + s) * DIMC + K) << 16) + (x << 8) + y];
    }
    __syncthreads();

    const int wave = tid >> 5;
    const int lane = tid & 31;
    const int half = lane >> 4;
    const int l16  = lane & 15;
    const int nt2  = wave >> 2;
    const int s    = (wave >> 1) & 1;
    const int mt   = wave & 1;
    const int m0   = mt * 16;

    const int P  = P0 + nt2 * 16 + l16;
    const int b  = P >> 16;
    const int rm = P & (HW - 1);
    const int x  = rm >> 8;
    const int y  = rm & 255;

    // A fragment from bf16 proj weights (row = out channel)
    const __bf16* __restrict__ Arow = pw + (size_t)(s * DIMC + m0 + l16) * DIMC;
    v8bf lo = *(const v8bf*)(Arow + (half << 4));
    v8bf hi = *(const v8bf*)(Arow + (half << 4) + 8);
    v16bf a = __builtin_shufflevector(lo, hi, 0, 1, 2, 3, 4, 5, 6, 7,
                                      8, 9, 10, 11, 12, 13, 14, 15);
    // B fragment: contiguous from transposed LDS tile
    v16bf bm = *(const v16bf*)(sT + ((nt2 * 2 + s) * 16 + l16) * 40 + (half << 4));

    v8f acc = {};
    acc = __builtin_amdgcn_wmma_f32_16x16x32_bf16(
        false, a, false, bm, (short)0, acc, false, false);

    const float* bias = s ? bias_ms : bias_pan;
#pragma unroll
    for (int r = 0; r < 8; ++r) {
        int oc = m0 + r + (half << 3);
        out[(size_t)s * (BATCH * DIMC * HW)
            + (((size_t)b * DIMC + oc) << 16) + (x << 8) + y] = acc[r] + bias[oc];
    }
}

// ---------------- host: launch pipeline ----------------
extern "C" void kernel_launch(void* const* d_in, const int* in_sizes, int n_in,
                              void* d_out, int out_size, void* d_ws, size_t ws_size,
                              hipStream_t stream)
{
    const float* x      = (const float*)d_in[0];
    const float* ms     = (const float*)d_in[1];
    const float* lpan   = (const float*)d_in[2];
    const float* pan    = (const float*)d_in[3];
    const float* sarr   = (const float*)d_in[4];
    const float* w_q    = (const float*)d_in[5];
    /* d_in[6] = w_kpan is dead: k_pan is overwritten by k_ms in the reference */
    const float* w_vpan = (const float*)d_in[7];
    const float* w_kvms = (const float*)d_in[8];
    const float* w_dep  = (const float*)d_in[9];
    const float* b_dep  = (const float*)d_in[10];
    const float* w_pp   = (const float*)d_in[11];
    const float* b_pp   = (const float*)d_in[12];
    const float* w_pm   = (const float*)d_in[13];
    const float* b_pm   = (const float*)d_in[14];
    float* out = (float*)d_out;

    // workspace layout (bf16 intermediates, 256B aligned)
    constexpr size_t U_elems   = (size_t)BATCH * CU * PADW * PADW;      // 6,789,528
    constexpr size_t U_bytes   = ((U_elems * 2 + 255) / 256) * 256;
    constexpr size_t W_elems   = (size_t)QKVC * KPAD;                   // 65,536
    constexpr size_t W_bytes   = W_elems * 2;                           // 131,072
    constexpr size_t PW_bytes  = 4096;                                  // 2*32*32 bf16
    constexpr size_t QKV_elems = (size_t)BATCH * QKVC * HW;             // 16,777,216
    constexpr size_t QKV_bytes = QKV_elems * 2;

    char* ws = (char*)d_ws;
    __bf16* U    = (__bf16*)(ws);
    __bf16* Wall = (__bf16*)(ws + U_bytes);
    __bf16* PW   = (__bf16*)(ws + U_bytes + W_bytes);
    __bf16* QKV  = (__bf16*)(ws + U_bytes + W_bytes + PW_bytes);
    __bf16* AO   = (__bf16*)(ws + U_bytes + W_bytes + PW_bytes + QKV_bytes);

    // 0) prep inputs/weights
    {
        int total = (int)U_elems;
        prep_U_kernel<<<(total + 255) / 256, 256, 0, stream>>>(x, ms, lpan, pan, sarr, U, total);
        int totW = (int)W_elems + 2048;
        prep_W_kernel<<<(totW + 255) / 256, 256, 0, stream>>>(w_q, w_vpan, w_kvms, w_pp, w_pm, Wall, PW);
    }
    // 1) q / v_pan / k_ms / v_ms : one implicit-GEMM conv, 128 out-ch, K=459 (padded 512)
    conv_gemm_kernel<<<(BATCH * HW) / NPB, 256, 0, stream>>>(U, Wall, QKV);
    // 2) depthwise-9 + attention + softmax (incl. scrambled k_ms branch)
    attn_kernel<<<(BATCH * HEADS * HW) / 256, 256, 0, stream>>>(QKV, w_dep, b_dep, AO);
    // 3) 1x1 projections -> d_out (x_pan then x_ms)
    proj_kernel<<<(BATCH * HW) / 32, 256, 0, stream>>>(AO, PW, b_pp, b_pm, out);

    (void)in_sizes; (void)n_in; (void)out_size; (void)ws_size;
}
